// Attention_52527450030781
// MI455X (gfx1250) — matrix operand order repaired and checked by reading, not compiled, not run on hardware
//
#include <hip/hip_runtime.h>
#include <stdint.h>

typedef __bf16 bf16;
typedef __bf16 bf16x16 __attribute__((ext_vector_type(16)));
typedef float  f32x8   __attribute__((ext_vector_type(8)));
typedef float  f32x4   __attribute__((ext_vector_type(4)));
typedef uint32_t u32x4 __attribute__((ext_vector_type(4)));
typedef int      i32x4 __attribute__((ext_vector_type(4)));
typedef int      i32x8 __attribute__((ext_vector_type(8)));

#if __has_builtin(__builtin_amdgcn_s_wait_tensorcnt)
#define WAIT_TENSORCNT0() __builtin_amdgcn_s_wait_tensorcnt(0)
#else
#define WAIT_TENSORCNT0() asm volatile("s_wait_tensorcnt 0x0" ::: "memory")
#endif

union Frag {
    bf16x16 v;
    u32x4   q[2];
};

// Native bf16 convert (lowers to packed v_fma_mixlo/mixhi_bf16 on gfx1250)
__device__ __forceinline__ bf16 f2bf(float f) { return (bf16)f; }

// A-matrix 16x32 bf16 fragment: half 0 -> K {0..7, 16..23}; half 1 -> K {8..15, 24..31}
__device__ __forceinline__ bf16x16 load_frag_a(const bf16* row, int half) {
    Frag f;
    f.q[0] = *reinterpret_cast<const u32x4*>(row + half * 8);
    f.q[1] = *reinterpret_cast<const u32x4*>(row + 16 + half * 8);
    return f.v;
}

// B-matrix 32x16 bf16 fragment (stored as rows of B^T): half 0 -> K {0..7, 16..23}; half 1 -> K {8..15, 24..31}
__device__ __forceinline__ bf16x16 load_frag_b(const bf16* row, int half) {
    Frag f;
    f.q[0] = *reinterpret_cast<const u32x4*>(row + half * 8);
    f.q[1] = *reinterpret_cast<const u32x4*>(row + 16 + half * 8);
    return f.v;
}

__device__ __forceinline__ f32x8 wmma_bf16(bf16x16 a, bf16x16 b, f32x8 c) {
    return __builtin_amdgcn_wmma_f32_16x16x32_bf16(false, a, false, b, (short)0, c, false, false);
}

// 2D bf16 tile load via Tensor Data Mover (TENSOR_LOAD_TO_LDS):
// rows x row_elems from global (row stride row_stride_elems) -> contiguous LDS.
// D# layout per CDNA5 ISA 8.3/8.4; 6-arg builtin on this toolchain.
__device__ __forceinline__ void tdm_load_2d(unsigned lds_addr, const bf16* gptr,
                                            unsigned rows, unsigned row_elems,
                                            unsigned row_stride_elems) {
    unsigned long long ga = (unsigned long long)(uintptr_t)gptr;
    u32x4 g0;
    g0[0] = 1u;                                   // count=1, user descriptor
    g0[1] = lds_addr;                             // lds_addr       [63:32]
    g0[2] = (unsigned)(ga & 0xFFFFFFFFu);         // global_addr lo [95:64]
    g0[3] = (unsigned)((ga >> 32) & 0x1FFFFFFu)   // global_addr hi [120:96]
          | (2u << 30);                           // type=2 (image) [127:126]
    i32x8 g1;
    g1[0] = (int)(1u << 16);                      // wg_mask=0, data_size=1 (2B)
    g1[1] = (int)(row_elems << 16);               // tensor_dim0 lo  (bits 79:48)
    g1[2] = (int)((row_elems >> 16) | (rows << 16));   // dim0 hi | tensor_dim1 lo
    g1[3] = (int)((rows >> 16) | (row_elems << 16));   // dim1 hi | tile_dim0
    g1[4] = (int)rows;                            // tile_dim1 | tile_dim2=0
    g1[5] = (int)row_stride_elems;                // tensor_dim0_stride lo
    g1[6] = 0;
    g1[7] = 0;
    i32x4 z4 = {0, 0, 0, 0};
    i32x8 z8 = {0, 0, 0, 0, 0, 0, 0, 0};
    __builtin_amdgcn_tensor_load_to_lds(g0, g1, z4, z4, z8, 0);
}

// ---------------------------------------------------------------------------
// Kernel 1: QKV projection. C[8192,3072] = X[8192,1024] @ Wqkv[1024,3072]
// Block tile 128x64, BK=64 -> 8 WMMA per wave between barriers.
// Epilogue -> bf16 qkv ws [3][b=4][h=16][n=2048][d=64], q pre-scaled by 0.125.
// Grid: (48,64), block 256 threads (8 wave32s).
// ---------------------------------------------------------------------------
__global__ __launch_bounds__(256) void qkv_gemm(const float* __restrict__ X,
                                                const float* __restrict__ W,
                                                bf16* __restrict__ qkv) {
    __shared__ bf16 As[128][64];  // [m][k] 16KB
    __shared__ bf16 Bs[64][64];   // [n][k]  8KB (transposed stage)

    const int t    = threadIdx.x;
    const int wave = t >> 5;      // M tile (0..7)
    const int lane = t & 31;
    const int half = lane >> 4;
    const int l16  = lane & 15;

    const int m0 = blockIdx.y * 128;
    const int n0 = blockIdx.x * 64;

    f32x8 acc[4] = {{}, {}, {}, {}};

    for (int k0 = 0; k0 < 1024; k0 += 64) {
        {   // stage A (128x64 f32 -> bf16), 128B per thread
            int row = t >> 1;
            int cb  = (t & 1) * 32;
            const float* g = X + (size_t)(m0 + row) * 1024 + k0 + cb;
#pragma unroll
            for (int c = 0; c < 8; ++c) {
                f32x4 a = *reinterpret_cast<const f32x4*>(g + c * 4);
#pragma unroll
                for (int j = 0; j < 4; ++j) As[row][cb + c * 4 + j] = f2bf(a[j]);
            }
        }
        {   // stage B transposed (64k x 64n f32 -> Bs[n][k] bf16)
            int kk = t >> 2;
            int nb = (t & 3) * 16;
            const float* g = W + (size_t)(k0 + kk) * 3072 + n0 + nb;
#pragma unroll
            for (int c = 0; c < 4; ++c) {
                f32x4 b = *reinterpret_cast<const f32x4*>(g + c * 4);
#pragma unroll
                for (int j = 0; j < 4; ++j) Bs[nb + c * 4 + j][kk] = f2bf(b[j]);
            }
        }
        __syncthreads();

#pragma unroll
        for (int ks = 0; ks < 2; ++ks) {
            bf16x16 a = load_frag_a(&As[wave * 16 + l16][ks * 32], half);
#pragma unroll
            for (int nt = 0; nt < 4; ++nt) {
                bf16x16 b = load_frag_b(&Bs[nt * 16 + l16][ks * 32], half);
                acc[nt] = wmma_bf16(a, b, acc[nt]);
            }
        }
        __syncthreads();
    }

    // Epilogue: scatter into [which][b][h][n][d] bf16 (all constant per block)
    const int which = n0 >> 10;
    const int h     = (n0 & 1023) >> 6;
    const int b     = m0 >> 11;
    const int nbase = (m0 & 2047) + wave * 16 + half * 8;
    const float scl = (which == 0) ? 0.125f : 1.0f;
    bf16* dst = qkv + (size_t)which * (size_t)(8192 * 1024)
                    + ((size_t)(b * 16 + h) * 2048) * 64;
#pragma unroll
    for (int nt = 0; nt < 4; ++nt) {
        int d0 = nt * 16 + l16;
#pragma unroll
        for (int r = 0; r < 8; ++r)
            dst[(size_t)(nbase + r) * 64 + d0] = f2bf(acc[nt][r] * scl);
    }
}

// ---------------------------------------------------------------------------
// Kernel 2: flash attention over one (b,h), 128 q-rows per block (16 per wave).
// Double-buffered K/V tiles; K staged via TDM (wave 0 issues, TENSORcnt),
// next tile's DMA overlapped with WMMA compute; one barrier per tile.
// Grid: (16,64), block 256 threads.
// ---------------------------------------------------------------------------
__global__ __launch_bounds__(256) void attn_kernel(const bf16* __restrict__ qkv,
                                                   bf16* __restrict__ attn_out) {
    __shared__ bf16 Ks[2][64][64];   // [buf][key][d]  16KB
    __shared__ bf16 Vt[2][64][64];   // [buf][d][key]  16KB
    __shared__ bf16 Ps[8][16][64];   // per-wave P staging [m][key] 16KB

    const int t    = threadIdx.x;
    const int wave = t >> 5;
    const int lane = t & 31;
    const int half = lane >> 4;
    const int l16  = lane & 15;

    const int bh = blockIdx.y;
    const int q0 = blockIdx.x * 128 + wave * 16;

    const bf16* qbase = qkv + (size_t)bh * 2048 * 64;
    const bf16* kbase = qkv + (size_t)1 * (8192 * 1024) + (size_t)bh * 2048 * 64;
    const bf16* vbase = qkv + (size_t)2 * (8192 * 1024) + (size_t)bh * 2048 * 64;

    bf16x16 qf0 = load_frag_a(qbase + (size_t)(q0 + l16) * 64 + 0,  half);
    bf16x16 qf1 = load_frag_a(qbase + (size_t)(q0 + l16) * 64 + 32, half);

    auto stage_tile = [&](int kt, int buf) {
        if (wave == 0)   // one TDM op: 64x64 bf16 row-major K tile
            tdm_load_2d((unsigned)(uintptr_t)&Ks[buf][0][0],
                        kbase + (size_t)kt * 64, 64, 64, 64);
        {   // V tile transposed (manual: TDM cannot transpose)
            int row = t >> 2, db = (t & 3) * 16;
            const bf16* g = vbase + (size_t)(kt + row) * 64 + db;
#pragma unroll
            for (int j = 0; j < 16; ++j) Vt[buf][db + j][row] = g[j];
        }
    };

    f32x8 o0 = {}, o1 = {}, o2 = {}, o3 = {};
    float mrun[8], lrun[8];
#pragma unroll
    for (int r = 0; r < 8; ++r) { mrun[r] = -3.0e38f; lrun[r] = 0.0f; }

    stage_tile(0, 0);    // prologue
    int cur = 0;

    for (int kt = 0; kt < 2048; kt += 64) {
        if (wave == 0) WAIT_TENSORCNT0();
        __syncthreads();                       // buf[cur] published

        if (kt + 64 < 2048)                    // overlap next tile's DMA/stores
            stage_tile(kt + 64, cur ^ 1);

        // S = Q @ K^T : 16 rows x 64 keys, accumulated over d=64 (2 k-steps)
        f32x8 s0 = {}, s1 = {}, s2 = {}, s3 = {};
#pragma unroll
        for (int s = 0; s < 2; ++s) {
            bf16x16 a = (s == 0) ? qf0 : qf1;
            s0 = wmma_bf16(a, load_frag_b(&Ks[cur][ 0 + l16][s * 32], half), s0);
            s1 = wmma_bf16(a, load_frag_b(&Ks[cur][16 + l16][s * 32], half), s1);
            s2 = wmma_bf16(a, load_frag_b(&Ks[cur][32 + l16][s * 32], half), s2);
            s3 = wmma_bf16(a, load_frag_b(&Ks[cur][48 + l16][s * 32], half), s3);
        }

        // Online softmax (row values live across the 16 lanes of this half-wave)
#pragma unroll
        for (int r = 0; r < 8; ++r) {
            float v0 = s0[r], v1 = s1[r], v2 = s2[r], v3 = s3[r];
            float mx = fmaxf(fmaxf(v0, v1), fmaxf(v2, v3));
#pragma unroll
            for (int off = 8; off >= 1; off >>= 1) mx = fmaxf(mx, __shfl_xor(mx, off, 16));
            float mnew = fmaxf(mrun[r], mx);
            float al   = __expf(mrun[r] - mnew);
            float p0 = __expf(v0 - mnew), p1 = __expf(v1 - mnew);
            float p2 = __expf(v2 - mnew), p3 = __expf(v3 - mnew);
            float ps = p0 + p1 + p2 + p3;
#pragma unroll
            for (int off = 8; off >= 1; off >>= 1) ps += __shfl_xor(ps, off, 16);
            lrun[r] = lrun[r] * al + ps;
            mrun[r] = mnew;
            o0[r] *= al; o1[r] *= al; o2[r] *= al; o3[r] *= al;
            int mrow = r + half * 8;
            Ps[wave][mrow][ 0 + l16] = f2bf(p0);
            Ps[wave][mrow][16 + l16] = f2bf(p1);
            Ps[wave][mrow][32 + l16] = f2bf(p2);
            Ps[wave][mrow][48 + l16] = f2bf(p3);
        }

        // O += P @ V  (same-wave LDS RAW: DS ops in-order within a wave)
#pragma unroll
        for (int s = 0; s < 2; ++s) {
            bf16x16 pa = load_frag_a(&Ps[wave][l16][s * 32], half);
            o0 = wmma_bf16(pa, load_frag_b(&Vt[cur][ 0 + l16][s * 32], half), o0);
            o1 = wmma_bf16(pa, load_frag_b(&Vt[cur][16 + l16][s * 32], half), o1);
            o2 = wmma_bf16(pa, load_frag_b(&Vt[cur][32 + l16][s * 32], half), o2);
            o3 = wmma_bf16(pa, load_frag_b(&Vt[cur][48 + l16][s * 32], half), o3);
        }
        cur ^= 1;
    }

    // Normalize and write [b*2048+n][h*64+d] bf16
    const int b = bh >> 4, h = bh & 15;
#pragma unroll
    for (int r = 0; r < 8; ++r) {
        float inv = 1.0f / lrun[r];
        int qn = q0 + r + half * 8;
        size_t base = ((size_t)(b * 2048 + qn)) * 1024 + h * 64;
        attn_out[base +  0 + l16] = f2bf(o0[r] * inv);
        attn_out[base + 16 + l16] = f2bf(o1[r] * inv);
        attn_out[base + 32 + l16] = f2bf(o2[r] * inv);
        attn_out[base + 48 + l16] = f2bf(o3[r] * inv);
    }
}

// ---------------------------------------------------------------------------
// Kernel 3: output projection. out[8192,1024] = attn(bf16) @ Wout[1024,1024]
// Block tile 128x64, BK=64; A tile (bf16) staged via TDM each k-step.
// Grid: (16,64), block 256 threads.
// ---------------------------------------------------------------------------
__global__ __launch_bounds__(256) void out_gemm(const bf16* __restrict__ A,
                                                const float* __restrict__ W,
                                                float* __restrict__ C) {
    __shared__ bf16 As[128][64];  // 16KB
    __shared__ bf16 Bs[64][64];   //  8KB

    const int t    = threadIdx.x;
    const int wave = t >> 5;      // M tile (0..7)
    const int lane = t & 31;
    const int half = lane >> 4;
    const int l16  = lane & 15;

    const int m0 = blockIdx.y * 128;
    const int n0 = blockIdx.x * 64;

    f32x8 acc[4] = {{}, {}, {}, {}};

    for (int k0 = 0; k0 < 1024; k0 += 64) {
        if (wave == 0)   // 128x64 bf16 tile, row stride 1024 elems, via TDM
            tdm_load_2d((unsigned)(uintptr_t)&As[0][0],
                        A + (size_t)m0 * 1024 + k0, 128, 64, 1024);
        {   // stage B transposed (f32 -> bf16)
            int kk = t >> 2;
            int nb = (t & 3) * 16;
            const float* g = W + (size_t)(k0 + kk) * 1024 + n0 + nb;
#pragma unroll
            for (int c = 0; c < 4; ++c) {
                f32x4 b = *reinterpret_cast<const f32x4*>(g + c * 4);
#pragma unroll
                for (int j = 0; j < 4; ++j) Bs[nb + c * 4 + j][kk] = f2bf(b[j]);
            }
        }
        if (wave == 0) WAIT_TENSORCNT0();
        __syncthreads();

#pragma unroll
        for (int ks = 0; ks < 2; ++ks) {
            bf16x16 a = load_frag_a(&As[wave * 16 + l16][ks * 32], half);
#pragma unroll
            for (int nt = 0; nt < 4; ++nt) {
                bf16x16 b = load_frag_b(&Bs[nt * 16 + l16][ks * 32], half);
                acc[nt] = wmma_bf16(a, b, acc[nt]);
            }
        }
        __syncthreads();
    }

    const int mr = m0 + wave * 16 + half * 8;
#pragma unroll
    for (int nt = 0; nt < 4; ++nt) {
        int nc = n0 + nt * 16 + l16;
#pragma unroll
        for (int r = 0; r < 8; ++r)
            C[(size_t)(mr + r) * 1024 + nc] = acc[nt][r];
    }
}

// ---------------------------------------------------------------------------
extern "C" void kernel_launch(void* const* d_in, const int* in_sizes, int n_in,
                              void* d_out, int out_size, void* d_ws, size_t ws_size,
                              hipStream_t stream) {
    const float* x     = (const float*)d_in[0];   // [4,2048,1024]
    const float* w_qkv = (const float*)d_in[1];   // [1024,3072]
    const float* w_out = (const float*)d_in[2];   // [1024,1024]
    float*       out   = (float*)d_out;           // [4,2048,1024]

    bf16* qkv  = (bf16*)d_ws;                              // 3 * 8192*1024 bf16
    bf16* attn = qkv + (size_t)3 * (size_t)(8192 * 1024);  // 8192*1024 bf16

    qkv_gemm   <<<dim3(48, 64), 256, 0, stream>>>(x, w_qkv, qkv);
    attn_kernel<<<dim3(16, 64), 256, 0, stream>>>(qkv, attn);
    out_gemm   <<<dim3(16, 64), 256, 0, stream>>>(attn, w_out, out);
}
